// Cell2SeqWrapper_61366492725886
// MI455X (gfx1250) — matrix operand order repaired and checked by reading, not compiled, not run on hardware
//
#include <hip/hip_runtime.h>
#include <hip/hip_bf16.h>

// GRU: B=8, T=4096, D=256, H=512.
// Phase 1: gx = x @ W_ih^T + b_ih      (WMMA bf16 GEMM, memory-bound ~10us)
// Phase 2: sequential scan, W_hh partitioned across 8 persistent WGs,
//          each slice LDS-resident in bf16 (192KB of the 320KB WGP LDS).
//          h exchanged as bf16 via async-DMA into LDS; fp32 h kept in VGPRs.

#define BB   8
#define TT   4096
#define DD   256
#define HH   512
#define GG3  (3 * HH)      // 1536
#define NWG  8             // workgroups in the scan kernel
#define WGH  64            // hidden indices owned per WG (8*64 = 512)

typedef __attribute__((ext_vector_type(16))) __bf16 v16bf;
typedef __attribute__((ext_vector_type(8)))  __bf16 v8bf;
typedef __attribute__((ext_vector_type(8)))  float  v8f;
typedef __attribute__((ext_vector_type(4)))  int    v4i;

typedef __attribute__((address_space(1))) v4i gv4i;   // global int4
typedef __attribute__((address_space(3))) v4i lv4i;   // LDS int4

#if __has_builtin(__builtin_amdgcn_global_load_async_to_lds_b128) && \
    __has_builtin(__builtin_amdgcn_s_wait_asynccnt)
#define HAVE_ASYNC_LDS 1
#else
#define HAVE_ASYNC_LDS 0
#endif

static __device__ __forceinline__ v8f wmma_bf16(v16bf a, v16bf b, v8f c) {
  // (neg_a, A, neg_b, B, c_mod, C, reuse_a, reuse_b)
  return __builtin_amdgcn_wmma_f32_16x16x32_bf16(false, a, false, b, (short)0, c,
                                                 false, false);
}

// A fragment (16x32 bf16, M rows x K): lane L holds row
// M = L&15; elements e <-> K = 16*half + e (contiguous 16).
static __device__ __forceinline__ v16bf a_frag_from_f32(const float* row, int half) {
  const float* p0 = row + 16 * half;
  const float* p1 = row + 16 * half + 8;
  v16bf a;
#pragma unroll
  for (int i = 0; i < 8; ++i) { a[i] = (__bf16)p0[i]; a[8 + i] = (__bf16)p1[i]; }
  return a;
}

// B fragment (32x16 bf16, K x N): lane L holds column N = L&15 (a row of the
// N-major weight matrix); elements e <-> K = 16*half + e (contiguous 16).
static __device__ __forceinline__ v16bf b_frag_from_f32(const float* row, int half) {
  const float* p = row + 16 * half;
  v16bf b;
#pragma unroll
  for (int i = 0; i < 16; ++i) b[i] = (__bf16)p[i];
  return b;
}

static __device__ __forceinline__ v16bf a_frag_from_lds(const __bf16* base, int half) {
  v8bf lo = *(const v8bf*)(base + 16 * half);
  v8bf hi = *(const v8bf*)(base + 16 * half + 8);
  return __builtin_shufflevector(lo, hi, 0, 1, 2, 3, 4, 5, 6, 7,
                                 8, 9, 10, 11, 12, 13, 14, 15);
}

// ---------------------------------------------------------------------------
// Phase 0: reset the grid-barrier counter (graph-replay safe / deterministic).
// ---------------------------------------------------------------------------
__global__ void init_kernel(unsigned* bar) {
  if (threadIdx.x == 0) *bar = 0u;
}

// ---------------------------------------------------------------------------
// Phase 1: gx[M=32768][N=1536] = x[M][K=256] @ W_ih[N][K]^T + b_ih
// Grid (512, 12), 256 threads = 8 waves; wave = 16M x 64N, K-loop 8x32.
// ---------------------------------------------------------------------------
__global__ void __launch_bounds__(256)
gx_gemm_kernel(const float* __restrict__ x, const float* __restrict__ W_ih,
               const float* __restrict__ b_ih, float* __restrict__ gx) {
  const int tid  = threadIdx.x;
  const int w    = tid >> 5;
  const int lane = tid & 31;
  const int half = lane >> 4;
  const int l16  = lane & 15;

  const int m_base = blockIdx.x * 64 + (w & 3) * 16;
  const int n_base = blockIdx.y * 128 + (w >> 2) * 64;

  const float* arow = x + (size_t)(m_base + l16) * DD;

  v8f acc[4] = {};
#pragma unroll
  for (int k0 = 0; k0 < DD; k0 += 32) {
    v16bf a = a_frag_from_f32(arow + k0, half);
#pragma unroll
    for (int nt = 0; nt < 4; ++nt) {
      const float* brow = W_ih + (size_t)(n_base + nt * 16 + l16) * DD + k0;
      v16bf b = b_frag_from_f32(brow, half);
      acc[nt] = wmma_bf16(a, b, acc[nt]);   // A reused across 4 N-tiles
    }
  }
  // C/D layout: lane L, vgpr v -> M = v + 8*half, N = L&15.
#pragma unroll
  for (int nt = 0; nt < 4; ++nt) {
    const int n = n_base + nt * 16 + l16;
    const float bias = b_ih[n];
#pragma unroll
    for (int v = 0; v < 8; ++v) {
      const int m = m_base + v + 8 * half;
      gx[(size_t)m * GG3 + n] = acc[nt][v] + bias;
    }
  }
}

// ---------------------------------------------------------------------------
// Grid barrier across the 8 persistent scan workgroups (monotonic counter).
// ---------------------------------------------------------------------------
static __device__ __forceinline__ void grid_sync(unsigned* cnt, unsigned target) {
  __threadfence();            // release this thread's h stores to device scope
  __syncthreads();
  if (threadIdx.x == 0) {
    __hip_atomic_fetch_add(cnt, 1u, __ATOMIC_ACQ_REL, __HIP_MEMORY_SCOPE_AGENT);
    while (__hip_atomic_load(cnt, __ATOMIC_ACQUIRE, __HIP_MEMORY_SCOPE_AGENT) < target)
      __builtin_amdgcn_s_sleep(2);
  }
  __syncthreads();
  __threadfence();            // acquire other WGs' h stores
}

// ---------------------------------------------------------------------------
// Phase 2: persistent scan. 8 WGs x 128 threads (4 waves).
// WG `wg` owns hidden indices [wg*64, wg*64+64): its r/z/n rows of W_hh
// (3*64*512 bf16 = 192KB) stay in LDS for all 4096 steps. h is broadcast as a
// double-buffered bf16 global buffer (8KB, L2-resident) and pulled into LDS
// with GLOBAL_LOAD_ASYNC_TO_LDS_B128 each step; the fp32 copy of each lane's
// own h values never leaves VGPRs. Wave w: one 16-col tile per gate.
// ---------------------------------------------------------------------------
__global__ void __launch_bounds__(128)
gru_scan_kernel(const float* __restrict__ gx, const float* __restrict__ W_hh,
                const float* __restrict__ b_hh, const float* __restrict__ h0,
                float* __restrict__ ys, float* __restrict__ hlast,
                __bf16* __restrict__ hbb0, __bf16* __restrict__ hbb1,
                unsigned* __restrict__ bar) {
  extern __shared__ char smem[];
  __bf16* Wl = (__bf16*)smem;                              // [3*WGH][HH] bf16
  __bf16* hA = (__bf16*)(smem + (size_t)3 * WGH * HH * 2); // [16][HH] bf16 (A staging)

  const int wg   = blockIdx.x;
  const int tid  = threadIdx.x;
  const int w    = tid >> 5;
  const int lane = tid & 31;
  const int half = lane >> 4;
  const int l16  = lane & 15;

  // ---- one-time: W_hh slice -> LDS bf16. idx = ((g*64 + r)*512 + k). ----
  for (int idx = tid; idx < 3 * WGH * HH; idx += 128) {
    const int g = idx / (WGH * HH);
    const int r = (idx / HH) % WGH;
    const int k = idx % HH;
    Wl[idx] = (__bf16)W_hh[(size_t)(g * HH + wg * WGH + r) * HH + k];
  }
  // zero the M=8..15 padding rows of the A staging buffer (never rewritten)
  for (int idx = tid; idx < 8 * HH; idx += 128) hA[8 * HH + idx] = (__bf16)0.0f;
  // seed the bf16 h broadcast buffer from h0 (this WG's hidden slice)
  for (int idx = tid; idx < BB * WGH; idx += 128) {
    const int m = idx / WGH, j = wg * WGH + (idx % WGH);
    hbb0[m * HH + j] = (__bf16)h0[m * HH + j];
  }

  // per-lane gate biases + register-resident fp32 h for this lane's column
  const int jg = wg * WGH + w * 16 + l16;
  const float br = b_hh[0 * HH + jg];
  const float bz = b_hh[1 * HH + jg];
  const float bn = b_hh[2 * HH + jg];
  float hprev[8];
#pragma unroll
  for (int v = 0; v < 8; ++v) hprev[v] = h0[v * HH + jg];

  grid_sync(bar, NWG * 1u);

  const __bf16* aBase  = hA + (size_t)l16 * HH;
  const __bf16* bBase0 = Wl + (size_t)(0 * WGH + w * 16 + l16) * HH;
  const __bf16* bBase1 = Wl + (size_t)(1 * WGH + w * 16 + l16) * HH;
  const __bf16* bBase2 = Wl + (size_t)(2 * WGH + w * 16 + l16) * HH;

  for (int t = 0; t < TT; ++t) {
    const __bf16* hc     = (t & 1) ? hbb1 : hbb0;
    __bf16*       hn_buf = (t & 1) ? hbb0 : hbb1;

    // ---- stage h broadcast (8x512 bf16 = 8KB) into LDS rows 0..7 ----
#if HAVE_ASYNC_LDS
    {
#pragma unroll
      for (int i = 0; i < 4; ++i) {
        const size_t off = (size_t)(tid + i * 128) * 16;   // 128 thr * 4 * 16B = 8KB
        __builtin_amdgcn_global_load_async_to_lds_b128(
            (gv4i*)((const char*)hc + off),
            (lv4i*)((char*)hA + off),
            0, 0);
      }
      __builtin_amdgcn_s_wait_asynccnt(0);
    }
#else
    {
      const unsigned* s = (const unsigned*)hc;   // 2048 dwords
      unsigned* d = (unsigned*)hA;
      for (int i = tid; i < 2048; i += 128) d[i] = s[i];
    }
#endif
    __syncthreads();

    v8f accr, accz, accn;
#pragma unroll
    for (int v = 0; v < 8; ++v) { accr[v] = br; accz[v] = bz; accn[v] = bn; }

#pragma unroll 4
    for (int kk = 0; kk < HH / 32; ++kk) {
      const int K0 = kk * 32;
      v16bf a  = a_frag_from_lds(aBase + K0, half);
      v16bf b0 = *(const v16bf*)(bBase0 + K0 + 16 * half);
      accr = wmma_bf16(a, b0, accr);
      v16bf b1 = *(const v16bf*)(bBase1 + K0 + 16 * half);
      accz = wmma_bf16(a, b1, accz);
      v16bf b2 = *(const v16bf*)(bBase2 + K0 + 16 * half);
      accn = wmma_bf16(a, b2, accn);
    }

    // elementwise gates: valid batches live in lanes 0..15 (M = v, v<8=B)
    if (half == 0) {
      const int j = jg;
#pragma unroll
      for (int v = 0; v < 8; ++v) {
        const int m = v;
        const size_t gxo = ((size_t)m * TT + t) * GG3;
        const float xr = gx[gxo + 0 * HH + j];
        const float xz = gx[gxo + 1 * HH + j];
        const float xn = gx[gxo + 2 * HH + j];
        const float rg = 1.0f / (1.0f + __expf(-(xr + accr[v])));
        const float zg = 1.0f / (1.0f + __expf(-(xz + accz[v])));
        const float ng = tanhf(xn + rg * accn[v]);
        const float hv = (1.0f - zg) * ng + zg * hprev[v];
        hprev[v] = hv;
        hn_buf[m * HH + j] = (__bf16)hv;          // bf16 broadcast for next step's A
        ys[((size_t)m * TT + t) * HH + j] = hv;   // fp32 output
        if (t == TT - 1) hlast[m * HH + j] = hv;
        if (t + 1 < TT) {  // hide next step's gx latency behind the WMMA chain
          __builtin_prefetch(&gx[gxo + GG3 + 0 * HH + j], 0, 0);
          __builtin_prefetch(&gx[gxo + GG3 + 1 * HH + j], 0, 0);
          __builtin_prefetch(&gx[gxo + GG3 + 2 * HH + j], 0, 0);
        }
      }
    }

    // one barrier/step suffices with double-buffered h
    grid_sync(bar, (unsigned)(NWG * (t + 2)));
  }
}

// ---------------------------------------------------------------------------
extern "C" void kernel_launch(void* const* d_in, const int* in_sizes, int n_in,
                              void* d_out, int out_size, void* d_ws, size_t ws_size,
                              hipStream_t stream) {
  const float* x    = (const float*)d_in[0];
  const float* h0   = (const float*)d_in[1];
  const float* W_ih = (const float*)d_in[2];
  const float* W_hh = (const float*)d_in[3];
  const float* b_ih = (const float*)d_in[4];
  const float* b_hh = (const float*)d_in[5];

  float* ys    = (float*)d_out;                    // (B,T,H)
  float* hlast = ys + (size_t)BB * TT * HH;        // (B,H)

  // workspace: [0..255] barrier counter | two 8KB bf16 h buffers | gx (192MB)
  unsigned* bar = (unsigned*)d_ws;
  __bf16* hbb0 = (__bf16*)((char*)d_ws + 256);
  __bf16* hbb1 = hbb0 + BB * HH;
  float*  gx   = (float*)((char*)d_ws + 65536);

  hipLaunchKernelGGL(init_kernel, dim3(1), dim3(64), 0, stream, bar);
  hipLaunchKernelGGL(gx_gemm_kernel, dim3(512, 12), dim3(256), 0, stream,
                     x, W_ih, b_ih, gx);

  const size_t ldsBytes = (size_t)(3 * WGH * HH + 16 * HH) * 2;  // 212992 B
  (void)hipFuncSetAttribute(reinterpret_cast<const void*>(gru_scan_kernel),
                            hipFuncAttributeMaxDynamicSharedMemorySize,
                            (int)ldsBytes);
  hipLaunchKernelGGL(gru_scan_kernel, dim3(NWG), dim3(128), ldsBytes, stream,
                     gx, W_hh, b_hh, h0, ys, hlast, hbb0, hbb1, bar);
}
